// Sequence_69698729280009
// MI455X (gfx1250) — matrix operand order repaired and checked by reading, not compiled, not run on hardware
//
#include <hip/hip_runtime.h>
#include <hip/hip_bf16.h>

typedef __attribute__((ext_vector_type(16))) _Float16 v16h;
typedef __attribute__((ext_vector_type(8)))  _Float16 v8h;
typedef __attribute__((ext_vector_type(8)))  float    v8f;
typedef __attribute__((ext_vector_type(4)))  float    v4f;

// ---------------- problem geometry (from reference) ----------------
constexpr int H     = 51;        // hidden size
constexpr int G     = 4 * H;     // 204 gate outputs
constexpr int NP    = 208;       // gate dim padded to 13*16 WMMA N-tiles
constexpr int NT    = 13;        // number of 16-wide N tiles
constexpr int KP    = 64;        // K padded: [0..50]=h, [51]=1(bias), [52]=x, rest 0
constexpr int KBIAS = 51;        // augmented K column carrying the bias
constexpr int KX    = 52;        // augmented K column carrying x_t (layer 1 only)
constexpr int BATCH = 4096;
constexpr int T     = 512;
constexpr int MROWS = 16;        // batch rows per wave (one WMMA M-tile)

// ---------------- workspace layout (bytes) ----------------
constexpr size_t WS_WHH1 = 0;                               // f16 [208][64], k51=bias1, k52=W_ih1
constexpr size_t WS_WIH2 = WS_WHH1 + (size_t)NP * KP * 2;   // f16 [208][64], k51=bias2
constexpr size_t WS_WHH2 = WS_WIH2 + (size_t)NP * KP * 2;   // f16 [208][64], pad zero
constexpr size_t WS_WOUT = WS_WHH2 + (size_t)NP * KP * 2;   // f32 [64]
constexpr size_t WS_BOUT = WS_WOUT + (size_t)64 * 4;        // f32 [1]

// =================== prep: pad/convert/augment weights to WMMA-friendly f16 ===================
__global__ void lstm_prep(const float* __restrict__ Wih1, const float* __restrict__ Whh1,
                          const float* __restrict__ bih1, const float* __restrict__ bhh1,
                          const float* __restrict__ Wih2, const float* __restrict__ Whh2,
                          const float* __restrict__ bih2, const float* __restrict__ bhh2,
                          const float* __restrict__ Wout, const float* __restrict__ bout,
                          unsigned char* __restrict__ ws) {
    _Float16* w1 = (_Float16*)(ws + WS_WHH1);
    _Float16* w2 = (_Float16*)(ws + WS_WIH2);
    _Float16* w3 = (_Float16*)(ws + WS_WHH2);
    float* wo = (float*)(ws + WS_WOUT);
    float* bo = (float*)(ws + WS_BOUT);

    int idx = blockIdx.x * blockDim.x + threadIdx.x;
    int stride = gridDim.x * blockDim.x;

    // Layout: W[n][k], n = gate output (0..207), k = augmented hidden (0..63).
    // Bias and the rank-1 input term ride in the K-padding so WMMA starts from C = inline 0.
    for (int i = idx; i < NP * KP; i += stride) {
        int n = i / KP, k = i % KP;
        float a = 0.f, b = 0.f, c = 0.f;
        if (n < G) {
            if (k < H) {
                a = Whh1[n * H + k];
                b = Wih2[n * H + k];
                c = Whh2[n * H + k];
            } else if (k == KBIAS) {
                a = bih1[n] + bhh1[n];
                b = bih2[n] + bhh2[n];
            } else if (k == KX) {
                a = Wih1[n];
            }
        }
        w1[i] = (_Float16)a;
        w2[i] = (_Float16)b;
        w3[i] = (_Float16)c;
    }
    for (int k = idx; k < 64; k += stride) wo[k] = (k < H) ? Wout[k] : 0.f;
    if (idx == 0) bo[0] = bout[0];
}

// =================== device helpers ===================
__device__ __forceinline__ float sigf(float x) {
    return 1.0f / (1.0f + __expf(-x));
}
__device__ __forceinline__ float tanh_fast(float x) {
    // tanh(x) = 1 - 2/(exp(2x)+1)
    return 1.0f - 2.0f / (__expf(2.0f * x) + 1.0f);
}

__device__ __forceinline__ v8f wmma_f16(v16h a, v16h b, v8f c) {
    return __builtin_amdgcn_wmma_f32_16x16x32_f16(
        /*neg_a=*/false, a, /*neg_b=*/false, b,
        /*c_mod=*/(short)0, c, /*reuse_a=*/false, /*reuse_b=*/false);
}

// A-fragment (16x32 f16) from LDS hidden-state buffer [16 rows][KP halves].
// CDNA5 A layout: lanes 0-15 row M=lane, elems 0-7 -> K=kt*32+0..7, 8-15 -> +16..23;
// lanes 16-31 same rows, K offset +8.  Two contiguous b128 chunks per lane.
__device__ __forceinline__ v16h load_a(const _Float16* hbuf, int lane, int kt) {
    int m  = lane & 15;
    int k0 = kt * 32 + ((lane >> 4) << 3);
    const _Float16* p = hbuf + m * KP + k0;
    v8h lo = *(const v8h*)(p);
    v8h hi = *(const v8h*)(p + 16);
    return __builtin_shufflevector(lo, hi, 0,1,2,3,4,5,6,7,8,9,10,11,12,13,14,15);
}

// B-fragment (32x16 f16) from global weights stored [N=208][K=64] f16 (loop-invariant:
// the compiler hoists these into high VGPRs via s_set_vgpr_msb).
__device__ __forceinline__ v16h load_b(const _Float16* __restrict__ w, int lane, int nt, int kt) {
    int n  = nt * 16 + (lane & 15);
    int k0 = kt * 32 + ((lane >> 4) << 3);
    const _Float16* p = w + n * KP + k0;
    v8h lo = *(const v8h*)(p);
    v8h hi = *(const v8h*)(p + 16);
    return __builtin_shufflevector(lo, hi, 0,1,2,3,4,5,6,7,8,9,10,11,12,13,14,15);
}

// Gate spill, transposed layout gatesT[n][m] so each lane's 8 accumulator rows are
// contiguous: two ds_store_b128 instead of eight ds_store_b32.
__device__ __forceinline__ void store_gates(float* gatesT, v8f acc, int lane, int nt) {
    int nc = nt * 16 + (lane & 15);
    int mb = (lane >> 4) * 8;
    float* p = gatesT + nc * MROWS + mb;
    *(v4f*)(p)     = __builtin_shufflevector(acc, acc, 0, 1, 2, 3);
    *(v4f*)(p + 4) = __builtin_shufflevector(acc, acc, 4, 5, 6, 7);
}

// =================== main: one wave == 16 independent batch recurrences ===================
__global__ void __launch_bounds__(32)
lstm_main(const float* __restrict__ x, const int* __restrict__ futurep,
          const unsigned char* __restrict__ ws, float* __restrict__ out) {
    __shared__ float    gatesT[NP * MROWS];    // 13.3 KB, [n][m] transposed
    __shared__ _Float16 h1f[MROWS * KP];       // [0..50]=h1, [51]=1, [52]=x_t
    __shared__ _Float16 h2f[MROWS * KP];       // [0..50]=h2, rest 0
    __shared__ float    c1s[MROWS * H];        // [j*16+m] layout (matches gatesT)
    __shared__ float    c2s[MROWS * H];
    __shared__ float    wos[64];
    __shared__ float    ob[MROWS];

    const int lane = threadIdx.x;
    const int gr0  = blockIdx.x * MROWS;

    const _Float16* Whh1 = (const _Float16*)(ws + WS_WHH1);
    const _Float16* Wih2 = (const _Float16*)(ws + WS_WIH2);
    const _Float16* Whh2 = (const _Float16*)(ws + WS_WHH2);
    const float* wog = (const float*)(ws + WS_WOUT);
    const float  bo  = *(const float*)(ws + WS_BOUT);
    const int    TT  = T + *futurep;

    // init state; h1f gets the constant-1 bias column
    for (int i = lane; i < MROWS * KP; i += 32) { h1f[i] = (_Float16)0.f; h2f[i] = (_Float16)0.f; }
    for (int i = lane; i < MROWS * H;  i += 32) { c1s[i] = 0.f; c2s[i] = 0.f; }
    if (lane < MROWS) h1f[lane * KP + KBIAS] = (_Float16)1.0f;
    for (int i = lane; i < 64; i += 32) wos[i] = wog[i];
    __syncthreads();

    const v8f ZC = {};   // inline-0 C operand for the first WMMA of each tile

    for (int t = 0; t < TT; ++t) {
        // inject x_t into the augmented K column (autoregressive after T real steps)
        if (lane < MROWS) {
            float xr = (t < T) ? x[(size_t)t * BATCH + gr0 + lane] : ob[lane];
            h1f[lane * KP + KX] = (_Float16)xr;
        }
        __syncthreads();

        // ---- layer 1 gates: [h1|1|x] @ [W_hh1|b1|W_ih1]^T   (bias + input via K-aug)
        {
            v16h a0 = load_a(h1f, lane, 0);
            v16h a1 = load_a(h1f, lane, 1);
#pragma unroll
            for (int nt = 0; nt < NT; ++nt) {
                v8f acc = wmma_f16(a0, load_b(Whh1, lane, nt, 0), ZC);
                acc     = wmma_f16(a1, load_b(Whh1, lane, nt, 1), acc);
                store_gates(gatesT, acc, lane, nt);
            }
        }
        __syncthreads();

        // ---- layer 1 pointwise: i,f,g,o at lane-contiguous addresses
        for (int i = lane; i < MROWS * H; i += 32) {
            int m = i & 15, j = i >> 4;
            float ig = gatesT[i];
            float fg = gatesT[i + H * MROWS];
            float gg = gatesT[i + 2 * H * MROWS];
            float og = gatesT[i + 3 * H * MROWS];
            float c  = sigf(fg) * c1s[i] + sigf(ig) * tanh_fast(gg);
            c1s[i]   = c;
            h1f[m * KP + j] = (_Float16)(sigf(og) * tanh_fast(c));
        }
        __syncthreads();

        // ---- layer 2 gates: [h1|1|x] @ [W_ih2|b2|0]^T + [h2|0] @ [W_hh2|0]^T
        {
            v16h a0 = load_a(h1f, lane, 0);
            v16h a1 = load_a(h1f, lane, 1);
            v16h d0 = load_a(h2f, lane, 0);
            v16h d1 = load_a(h2f, lane, 1);
#pragma unroll
            for (int nt = 0; nt < NT; ++nt) {
                v8f acc = wmma_f16(a0, load_b(Wih2, lane, nt, 0), ZC);
                acc     = wmma_f16(a1, load_b(Wih2, lane, nt, 1), acc);
                acc     = wmma_f16(d0, load_b(Whh2, lane, nt, 0), acc);
                acc     = wmma_f16(d1, load_b(Whh2, lane, nt, 1), acc);
                store_gates(gatesT, acc, lane, nt);
            }
        }
        __syncthreads();

        // ---- layer 2 pointwise
        for (int i = lane; i < MROWS * H; i += 32) {
            int m = i & 15, j = i >> 4;
            float ig = gatesT[i];
            float fg = gatesT[i + H * MROWS];
            float gg = gatesT[i + 2 * H * MROWS];
            float og = gatesT[i + 3 * H * MROWS];
            float c  = sigf(fg) * c2s[i] + sigf(ig) * tanh_fast(gg);
            c2s[i]   = c;
            h2f[m * KP + j] = (_Float16)(sigf(og) * tanh_fast(c));
        }
        __syncthreads();

        // ---- projection: out = h2 @ W_out^T + b_out, split over lane pairs
        {
            int m = lane & 15, half = lane >> 4;
            float acc = 0.f;
            for (int j = half; j < H; j += 2) acc += (float)h2f[m * KP + j] * wos[j];
            acc += __shfl_xor(acc, 16);
            if (lane < MROWS) {
                float val = acc + bo;
                ob[m] = val;
                out[(size_t)(gr0 + m) * TT + t] = val;
            }
        }
        __syncthreads();
    }
}

// =================== host entry ===================
extern "C" void kernel_launch(void* const* d_in, const int* in_sizes, int n_in,
                              void* d_out, int out_size, void* d_ws, size_t ws_size,
                              hipStream_t stream) {
    (void)in_sizes; (void)n_in; (void)out_size; (void)ws_size;
    const float* x    = (const float*)d_in[0];
    const float* Wih1 = (const float*)d_in[1];
    const float* Whh1 = (const float*)d_in[2];
    const float* bih1 = (const float*)d_in[3];
    const float* bhh1 = (const float*)d_in[4];
    const float* Wih2 = (const float*)d_in[5];
    const float* Whh2 = (const float*)d_in[6];
    const float* bih2 = (const float*)d_in[7];
    const float* bhh2 = (const float*)d_in[8];
    const float* Wout = (const float*)d_in[9];
    const float* bout = (const float*)d_in[10];
    const int*   fut  = (const int*)d_in[11];
    float* out = (float*)d_out;
    unsigned char* ws = (unsigned char*)d_ws;

    lstm_prep<<<64, 256, 0, stream>>>(Wih1, Whh1, bih1, bhh1,
                                      Wih2, Whh2, bih2, bhh2,
                                      Wout, bout, ws);
    lstm_main<<<BATCH / MROWS, 32, 0, stream>>>(x, fut, ws, out);
}
